// SphericalChebPoolConcat_75883482185961
// MI455X (gfx1250) — matrix operand order repaired and checked by reading, not compiled, not run on hardware
//
#include <hip/hip_runtime.h>
#include <hip/hip_bf16.h>

// ---------------------------------------------------------------------------
// SphericalChebPoolConcat for MI455X (gfx1250, wave32, WMMA)
//
//   pool4          : x[786432,64] -> P0[196608,64]
//   conv1 (F=64)   : Chebyshev recursion in {P0,P1}, k-PAIR-fused WMMA GEMMs
//                    accumulating into H (stride 128, cols 0..63); final pair
//                    fuses +b1 / PReLU.
//   copy_concat    : H[:,64:128] = concat_data
//   conv2 (F=128)  : recursion in {H,P1}, pair-fused WMMA GEMMs into d_out;
//                    final pair fuses +b2 / PReLU.
//
// GEMMs: v_wmma_f32_16x16x32_bf16, fp32 accum. Weight k-pair slices staged to
// LDS pre-swizzled into B-fragment order. A sched_barrier (blocking DS+WMMA,
// letting VALU/VMEM cross) pins all 4 B-fragment ds_loads of a chunk before
// the 4 WMMAs so the dscnt waits stagger instead of serializing per-WMMA.
// SpMM / recursion kept fp32 (bandwidth-bound random gather).
// ---------------------------------------------------------------------------

#define V_IN   786432
#define V_POOL 196608
#define NNZ    9

// sched_barrier mask: allow VALU(0x2)/SALU(0x4)/VMEM(0x10|0x20|0x40) to cross,
// block WMMA(0x8) and DS(0x80|0x100|0x200).
#define SCHED_ALLOW_ALU_VMEM 0x76

typedef __attribute__((ext_vector_type(16))) __bf16 v16bf;
typedef __attribute__((ext_vector_type(8)))  float  v8f;

__device__ __forceinline__ float prelu_f(float x, float a) {
    return x > 0.0f ? x : a * x;
}

// -------------------- pool: mean over vertex blocks of 4 -------------------
__global__ __launch_bounds__(256)
void pool4_kernel(const float* __restrict__ x, float* __restrict__ xp) {
    long idx = (long)blockIdx.x * 256 + threadIdx.x;   // over V_POOL*16 float4s
    long v  = idx >> 4;
    int  c4 = (int)(idx & 15) * 4;
    const float* p = x + v * 4 * 64 + c4;
    float sx = 0.f, sy = 0.f, sz = 0.f, sw = 0.f;
#pragma unroll
    for (int i = 0; i < 4; ++i) {
        float4 t = *(const float4*)(p + (long)i * 64);
        sx += t.x; sy += t.y; sz += t.z; sw += t.w;
    }
    float4 o; o.x = 0.25f * sx; o.y = 0.25f * sy; o.z = 0.25f * sz; o.w = 0.25f * sw;
    *(float4*)(xp + v * 64 + c4) = o;
}

// -------------------- SpMM + Chebyshev combine -----------------------------
// Xout[r,:] = alpha * sum_j vals[r*9+j] * Xin[cols[r*9+j],:]  + beta * Xm2[r,:]
// One wave32 per row. Xout may alias Xm2 (in-place, same-row read-then-write).
template <int F>
__global__ __launch_bounds__(256)
void spmm_cheb_kernel(const float* __restrict__ Xin,
                      const float* __restrict__ Xm2,
                      float* __restrict__ Xout,
                      const int* __restrict__ cols,
                      const float* __restrict__ vals,
                      float alpha, float beta) {
    const int wave = threadIdx.x >> 5;
    const int lane = threadIdx.x & 31;
    const long r = (long)blockIdx.x * 8 + wave;
    const long e0 = r * NNZ;

    if (F == 64) {
        const int c2 = lane * 2;
        float sx = 0.f, sy = 0.f;
#pragma unroll
        for (int j = 0; j < NNZ; ++j) {
            const int   c = cols[e0 + j];
            const float v = vals[e0 + j];
            float2 xx = *(const float2*)(Xin + (long)c * 64 + c2);
            sx = fmaf(v, xx.x, sx);
            sy = fmaf(v, xx.y, sy);
        }
        float2 o;
        if (beta != 0.0f) {
            float2 p = *(const float2*)(Xm2 + r * 64 + c2);
            o.x = fmaf(alpha, sx, beta * p.x);
            o.y = fmaf(alpha, sy, beta * p.y);
        } else {
            o.x = alpha * sx; o.y = alpha * sy;
        }
        *(float2*)(Xout + r * 64 + c2) = o;
    } else {
        const int c4 = lane * 4;
        float sx = 0.f, sy = 0.f, sz = 0.f, sw = 0.f;
#pragma unroll
        for (int j = 0; j < NNZ; ++j) {
            const int   c = cols[e0 + j];
            const float v = vals[e0 + j];
            float4 xx = *(const float4*)(Xin + (long)c * 128 + c4);
            sx = fmaf(v, xx.x, sx);
            sy = fmaf(v, xx.y, sy);
            sz = fmaf(v, xx.z, sz);
            sw = fmaf(v, xx.w, sw);
        }
        float4 o;
        if (beta != 0.0f) {
            float4 p = *(const float4*)(Xm2 + r * 128 + c4);
            o.x = fmaf(alpha, sx, beta * p.x);
            o.y = fmaf(alpha, sy, beta * p.y);
            o.z = fmaf(alpha, sz, beta * p.z);
            o.w = fmaf(alpha, sw, beta * p.w);
        } else {
            o.x = alpha * sx; o.y = alpha * sy; o.z = alpha * sz; o.w = alpha * sw;
        }
        *(float4*)(Xout + r * 128 + c4) = o;
    }
}

// --------- k-pair fused WMMA GEMM:
//   OUT[v, 0..63] (+)= X0[v,:]*Wpair[0] + X1[v,:]*Wpair[1]   (v-major, stride os)
// mode bit0: pre-read OUT into the accumulator; bit1: fuse +bias / PReLU.
// 256 threads = 8 waves; wave computes a 16x64 strip with four f32 16x16
// accumulators; reduction = 2*FIN in chunks of 32 via v_wmma_f32_16x16x32_bf16.
template <int FIN>
__global__ __launch_bounds__(256)
void cheb_gemm2_kernel(const float* __restrict__ X0,
                       const float* __restrict__ X1,
                       const float* __restrict__ Wpair,  // [2,FIN,64] fp32
                       float* __restrict__ OUT,
                       int os,                           // OUT row stride
                       const float* __restrict__ bias,
                       const float* __restrict__ slope_p,
                       int mode) {
    const int HS = FIN * 64;
    __shared__ __attribute__((aligned(32))) __bf16 wlds[2 * FIN * 64];
    const int tid = threadIdx.x;
    // Stage + swizzle into B-fragment order (coalesced source reads).
    for (int s = tid; s < 2 * HS; s += 256) {
        const int half = s / HS;
        const int s2   = s - half * HS;
        const int k    = s2 >> 6;
        const int n    = s2 & 63;
        const int nt   = n >> 4;
        const int m    = n & 15;
        const int kc_i = k >> 5;
        const int kr   = k & 31;
        const int l    = ((kr >> 4) << 4) | m;
        const int j    = kr & 15;
        wlds[half * HS + (((((kc_i << 2) + nt) << 5) + l) << 4) + j] =
            (__bf16)Wpair[s];
    }
    __syncthreads();

    const int wave = tid >> 5;
    const int lane = tid & 31;
    const int m16  = lane & 15;
    const int hi   = lane >> 4;
    const long row0 = (long)blockIdx.x * 128 + wave * 16;
    const float* xr0 = X0 + (row0 + m16) * FIN;
    const float* xr1 = X1 + (row0 + m16) * FIN;

    v8f acc[4];
    if (mode & 1) {
#pragma unroll
        for (int nt = 0; nt < 4; ++nt)
#pragma unroll
            for (int i = 0; i < 8; ++i)
                acc[nt][i] = OUT[(row0 + i + 8 * hi) * os + nt * 16 + m16];
    } else {
#pragma unroll
        for (int nt = 0; nt < 4; ++nt) {
            v8f z = {0.f, 0.f, 0.f, 0.f, 0.f, 0.f, 0.f, 0.f};
            acc[nt] = z;
        }
    }

    // A fragment double buffer across chunks (reduce WMMA->VALU WAR stalls).
    v16bf abuf[2];
#pragma unroll
    for (int cc = 0; cc < 2 * (FIN / 32); ++cc) {
        const int half = cc / (FIN / 32);
        const int kc_i = cc - half * (FIN / 32);
        const float* xr = half ? xr1 : xr0;
        const int wbase = half * HS;

        // Build A fragment (16x32 bf16, contiguous k = 16*hi + j, same order as B) into alternating buffer.
        v16bf& a = abuf[cc & 1];
        const int kb0 = kc_i * 32 + hi * 16;
#pragma unroll
        for (int j = 0; j < 8; ++j) a[j]     = (__bf16)xr[kb0 + j];
#pragma unroll
        for (int j = 0; j < 8; ++j) a[8 + j] = (__bf16)xr[kb0 + 8 + j];

        // Load all 4 B fragments of this chunk, then fence: DS and WMMA may
        // not cross, VALU/VMEM may (next chunk's global loads can hoist).
        v16bf b0 = *(const v16bf*)&wlds[wbase + ((((kc_i << 2) + 0) << 5 | lane) << 4)];
        v16bf b1 = *(const v16bf*)&wlds[wbase + ((((kc_i << 2) + 1) << 5 | lane) << 4)];
        v16bf b2 = *(const v16bf*)&wlds[wbase + ((((kc_i << 2) + 2) << 5 | lane) << 4)];
        v16bf b3 = *(const v16bf*)&wlds[wbase + ((((kc_i << 2) + 3) << 5 | lane) << 4)];
        __builtin_amdgcn_sched_barrier(SCHED_ALLOW_ALU_VMEM);

        acc[0] = __builtin_amdgcn_wmma_f32_16x16x32_bf16(
            false, a, false, b0, (short)0, acc[0], false, false);
        acc[1] = __builtin_amdgcn_wmma_f32_16x16x32_bf16(
            false, a, false, b1, (short)0, acc[1], false, false);
        acc[2] = __builtin_amdgcn_wmma_f32_16x16x32_bf16(
            false, a, false, b2, (short)0, acc[2], false, false);
        acc[3] = __builtin_amdgcn_wmma_f32_16x16x32_bf16(
            false, a, false, b3, (short)0, acc[3], false, false);
    }

    if (mode & 2) {
        const float sl = slope_p[0];
#pragma unroll
        for (int nt = 0; nt < 4; ++nt) {
            const float bn = bias[nt * 16 + m16];
#pragma unroll
            for (int i = 0; i < 8; ++i)
                OUT[(row0 + i + 8 * hi) * os + nt * 16 + m16] =
                    prelu_f(acc[nt][i] + bn, sl);
        }
    } else {
#pragma unroll
        for (int nt = 0; nt < 4; ++nt)
#pragma unroll
            for (int i = 0; i < 8; ++i)
                OUT[(row0 + i + 8 * hi) * os + nt * 16 + m16] = acc[nt][i];
    }
}

// -------------------- copy concat_data into hcat cols 64..127 --------------
__global__ __launch_bounds__(256)
void copy_concat_kernel(const float* __restrict__ cdat,
                        float* __restrict__ hcat) {
    long idx = (long)blockIdx.x * 256 + threadIdx.x;  // over V_POOL*16 float4s
    long v  = idx >> 4;
    int  c4 = (int)(idx & 15) * 4;
    *(float4*)(hcat + v * 128 + 64 + c4) = *(const float4*)(cdat + v * 64 + c4);
}

// ---------------------------------------------------------------------------
extern "C" void kernel_launch(void* const* d_in, const int* in_sizes, int n_in,
                              void* d_out, int out_size, void* d_ws, size_t ws_size,
                              hipStream_t stream) {
    (void)in_sizes; (void)n_in; (void)out_size; (void)ws_size;

    const float* x     = (const float*)d_in[0];   // [1, V_IN, 64]
    const float* cdat  = (const float*)d_in[1];   // [1, V_POOL, 64]
    // d_in[2] = lap_rows (structure fixed: 9 nnz per row, row-sorted)
    const int*   cols  = (const int*)d_in[3];     // [E]
    const float* vals  = (const float*)d_in[4];   // [E]
    const float* w1    = (const float*)d_in[5];   // [8,64,64]
    const float* b1    = (const float*)d_in[6];   // [64]
    const float* a1    = (const float*)d_in[7];   // [1]
    const float* w2    = (const float*)d_in[8];   // [8,128,64]
    const float* b2    = (const float*)d_in[9];   // [64]
    const float* a2    = (const float*)d_in[10];  // [1]
    float*       out   = (float*)d_out;           // [V_POOL,64]

    // Workspace (floats): P0[V,64] | P1[V,128] | H[V,128]   (~252 MB)
    float* P0 = (float*)d_ws;
    float* P1 = P0 + (size_t)V_POOL * 64;
    float* H  = P1 + (size_t)V_POOL * 128;

    const int gPool = (V_POOL * 16) / 256;   // 12288
    const int gSpmm = V_POOL / 8;            // 24576
    const int gGemm = V_POOL / 128;          // 1536
    const int gCopy = (V_POOL * 16) / 256;   // 12288

    // ---------------- conv1 (F = 64): even x_k in P0, odd in P1 -------------
    pool4_kernel<<<gPool, 256, 0, stream>>>(x, P0);                       // x0
    spmm_cheb_kernel<64><<<gSpmm, 256, 0, stream>>>(P0, P0, P1, cols, vals,
                                                    1.0f, 0.0f);          // x1
    // pair (x0,x1): init H cols 0..63 (stride 128)
    cheb_gemm2_kernel<64><<<gGemm, 256, 0, stream>>>(P0, P1, w1, H, 128,
                                                     b1, a1, 0);
    for (int p = 1; p < 4; ++p) {
        // x_{2p}   = 2 L x_{2p-1} - x_{2p-2}   (P1 -> P0, in place)
        spmm_cheb_kernel<64><<<gSpmm, 256, 0, stream>>>(P1, P0, P0, cols, vals,
                                                        2.0f, -1.0f);
        // x_{2p+1} = 2 L x_{2p}   - x_{2p-1}   (P0 -> P1, in place)
        spmm_cheb_kernel<64><<<gSpmm, 256, 0, stream>>>(P0, P1, P1, cols, vals,
                                                        2.0f, -1.0f);
        // pair GEMM; last pair fuses +b1 / PReLU
        cheb_gemm2_kernel<64><<<gGemm, 256, 0, stream>>>(
            P0, P1, w1 + (size_t)p * 2 * 64 * 64, H, 128, b1, a1,
            (p == 3) ? 3 : 1);
    }
    // hcat cols 64..127 = concat_data
    copy_concat_kernel<<<gCopy, 256, 0, stream>>>(cdat, H);

    // ---------------- conv2 (F = 128): even x_k in H, odd in P1 -------------
    spmm_cheb_kernel<128><<<gSpmm, 256, 0, stream>>>(H, H, P1, cols, vals,
                                                     1.0f, 0.0f);         // x1
    cheb_gemm2_kernel<128><<<gGemm, 256, 0, stream>>>(H, P1, w2, out, 64,
                                                      b2, a2, 0);
    for (int p = 1; p < 4; ++p) {
        spmm_cheb_kernel<128><<<gSpmm, 256, 0, stream>>>(P1, H, H, cols, vals,
                                                         2.0f, -1.0f);
        spmm_cheb_kernel<128><<<gSpmm, 256, 0, stream>>>(H, P1, P1, cols, vals,
                                                         2.0f, -1.0f);
        cheb_gemm2_kernel<128><<<gGemm, 256, 0, stream>>>(
            H, P1, w2 + (size_t)p * 2 * 128 * 64, out, 64, b2, a2,
            (p == 3) ? 3 : 1);
    }
}
